// FCOS_49254684950802
// MI455X (gfx1250) — compile-verified, matrix operand order checked by reading, not run on hardware
//
#include <hip/hip_runtime.h>

// ---------------------------------------------------------------------------
// FCOS head for MI455X (gfx1250): implicit-GEMM 3x3 convs via
// v_wmma_f32_16x16x32_bf16, zero-padded NHWC bf16 activations, 32x32
// register tile per wave (4 independent accumulator chains, 2:1 load:wmma).
// ---------------------------------------------------------------------------

typedef __attribute__((ext_vector_type(16))) __bf16 v16bf;
typedef __attribute__((ext_vector_type(8)))  float  v8f;

__device__ __forceinline__ unsigned short f2bf(float f) {
    unsigned int u = __float_as_uint(f);
    unsigned int r = u + 0x7fffu + ((u >> 16) & 1u); // round-to-nearest-even
    return (unsigned short)(r >> 16);
}

// ---------------------------------------------------------------------------
// Zero-fill (16B granules) -- clears padded activation buffers.
// ---------------------------------------------------------------------------
__global__ void zero16_kernel(uint4* __restrict__ p, long long n16) {
    long long i = (long long)blockIdx.x * blockDim.x + threadIdx.x;
    if (i < n16) {
        uint4 z; z.x = 0; z.y = 0; z.z = 0; z.w = 0;
        p[i] = z;
    }
}

// ---------------------------------------------------------------------------
// Pack conv weights OIHW f32 -> [O][tap][ci] bf16  (tap = ky*3+kx, I = 256)
// ---------------------------------------------------------------------------
__global__ void pack_w_kernel(const float* __restrict__ w,
                              unsigned short* __restrict__ wp, int O) {
    int i = blockIdx.x * blockDim.x + threadIdx.x;
    int total = O * 2304;
    if (i >= total) return;
    int o   = i / 2304;
    int r   = i % 2304;
    int tap = r / 256;
    int ci  = r % 256;
    wp[i] = f2bf(w[(o * 256 + ci) * 9 + tap]);   // w[o][ci][ky][kx]
}

// ---------------------------------------------------------------------------
// NCHW f32 -> padded NHWC bf16: dst [4, H+2, W+2, 256], interior only.
// W is a power of two (128/64/32/16/8) -> pass wshift.
// ---------------------------------------------------------------------------
__global__ void nchw_to_nhwc_pad_kernel(const float* __restrict__ x,
                                        unsigned short* __restrict__ act,
                                        int H, int W, int wshift) {
    long long i = (long long)blockIdx.x * blockDim.x + threadIdx.x;
    long long HW = (long long)H * W;
    long long total = 4LL * 256 * HW;
    if (i >= total) return;
    int c = (int)(i & 255);
    long long rem = i >> 8;                 // b*HW + p
    int p = (int)(rem % HW);
    int b = (int)(rem / HW);
    int h = p >> wshift;
    int w = p & (W - 1);
    int PW = W + 2, PH = H + 2;
    size_t dst = ((size_t)(b * PH + h + 1) * PW + (w + 1)) * 256 + c;
    act[dst] = f2bf(x[((long long)(b * 256 + c)) * HW + p]);
}

// ---------------------------------------------------------------------------
// 3x3 SAME conv, implicit GEMM with WMMA bf16 -> f32.
//   act   : [4, H+2, W+2, 256] bf16, zero border
//   wpack : [Cout, 2304]       bf16  (k = tap*256 + ci)
//   out   : [4, HW, Cout]      f32   (+bias, optional relu)
// grid.x = 4 * ceil(HW/32), grid.y = ceil(Cout/128), block = 128 (4 waves).
// Wave computes a 32(spatial) x 32(outch) tile: 4 accumulator chains,
// A/B fragments each reused twice -> 2 b128 loads per WMMA, and 4
// independent WMMA chains to bury the RAW hazard shadow + load latency.
// ---------------------------------------------------------------------------
__global__ void conv3x3_wmma_kernel(const unsigned short* __restrict__ act,
                                    const unsigned short* __restrict__ wpack,
                                    const float* __restrict__ bias,
                                    float* __restrict__ out,
                                    int H, int W, int Cout, int relu) {
    const int HW = H * W;
    const int PW = W + 2;
    const int PH = H + 2;
    const int mtiles = (HW + 31) >> 5;     // 32-position supertiles

    int b    = blockIdx.x / mtiles;
    int mt   = blockIdx.x % mtiles;
    int wave = threadIdx.x >> 5;
    int lane = threadIdx.x & 31;
    int m    = lane & 15;
    int grp  = lane >> 4;

    int n0   = (blockIdx.y * 8 + wave * 2) * 16;   // two 16-wide N tiles

    int pos0 = mt * 32 + m;
    int pos1 = pos0 + 16;
    int p0c  = pos0 < HW ? pos0 : HW - 1;
    int p1c  = pos1 < HW ? pos1 : HW - 1;
    int h0 = p0c / W, w0 = p0c % W;
    int h1 = p1c / W, w1 = p1c % W;

    int na  = n0 + m;        int nac = na  < Cout ? na  : Cout - 1;
    int nb2 = n0 + 16 + m;   int nbc = nb2 < Cout ? nb2 : Cout - 1;

    const uint4* wr0 = reinterpret_cast<const uint4*>(wpack + (size_t)nac * 2304);
    const uint4* wr1 = reinterpret_cast<const uint4*>(wpack + (size_t)nbc * 2304);

    v8f acc00 = {}, acc01 = {}, acc10 = {}, acc11 = {};

    for (int dy = 0; dy < 3; ++dy) {
        // padded coords: input row = h + dy, col base = w (dx adds 0..2)
        const uint4* ar0 = reinterpret_cast<const uint4*>(
            act + ((size_t)(b * PH + h0 + dy) * PW + w0) * 256);
        const uint4* ar1 = reinterpret_cast<const uint4*>(
            act + ((size_t)(b * PH + h1 + dy) * PW + w1) * 256);
        const uint4* w0t = wr0 + dy * 96;   // 3 taps * 32 uint4
        const uint4* w1t = wr1 + dy * 96;
        if (dy < 2) {                        // prefetch next tap-row of weights
            __builtin_prefetch(w0t + 96, 0, 1);
            __builtin_prefetch(w1t + 96, 0, 1);
        }
#pragma unroll
        for (int dx = 0; dx < 3; ++dx) {
#pragma unroll
            for (int cb = 0; cb < 8; ++cb) {
                // A 16x32 bf16 layout: lane group g holds K runs
                // [cb*32+g*8, +8) and [cb*32+16+g*8, +8) for row M = lane%16.
                int o0 = dx * 32 + cb * 4 + grp;
                int o1 = o0 + 2;
                union { uint4 u[2]; v16bf v; } a0, a1, f0, f1;
                a0.u[0] = ar0[o0]; a0.u[1] = ar0[o1];
                a1.u[0] = ar1[o0]; a1.u[1] = ar1[o1];
                f0.u[0] = w0t[o0]; f0.u[1] = w0t[o1];
                f1.u[0] = w1t[o0]; f1.u[1] = w1t[o1];
                acc00 = __builtin_amdgcn_wmma_f32_16x16x32_bf16(
                    false, a0.v, false, f0.v, (short)0, acc00, false, false);
                acc01 = __builtin_amdgcn_wmma_f32_16x16x32_bf16(
                    false, a0.v, false, f1.v, (short)0, acc01, false, false);
                acc10 = __builtin_amdgcn_wmma_f32_16x16x32_bf16(
                    false, a1.v, false, f0.v, (short)0, acc10, false, false);
                acc11 = __builtin_amdgcn_wmma_f32_16x16x32_bf16(
                    false, a1.v, false, f1.v, (short)0, acc11, false, false);
            }
        }
    }

    // C/D layout: VGPR r -> M = r (lanes 0-15), M = r+8 (lanes 16-31); N = lane%16.
    float bva = (na  < Cout) ? bias[na]  : 0.f;
    float bvb = (nb2 < Cout) ? bias[nb2] : 0.f;
#pragma unroll
    for (int r = 0; r < 8; ++r) {
        int mm0 = mt * 32 + r + grp * 8;
        int mm1 = mm0 + 16;
        if (na < Cout) {
            if (mm0 < HW) {
                float v = acc00[r] + bva;
                if (relu) v = v > 0.f ? v : 0.f;
                out[((long long)b * HW + mm0) * Cout + na] = v;
            }
            if (mm1 < HW) {
                float v = acc10[r] + bva;
                if (relu) v = v > 0.f ? v : 0.f;
                out[((long long)b * HW + mm1) * Cout + na] = v;
            }
        }
        if (nb2 < Cout) {
            if (mm0 < HW) {
                float v = acc01[r] + bvb;
                if (relu) v = v > 0.f ? v : 0.f;
                out[((long long)b * HW + mm0) * Cout + nb2] = v;
            }
            if (mm1 < HW) {
                float v = acc11[r] + bvb;
                if (relu) v = v > 0.f ? v : 0.f;
                out[((long long)b * HW + mm1) * Cout + nb2] = v;
            }
        }
    }
}

// ---------------------------------------------------------------------------
// GroupNorm stats: one block per (b, group); y: [4, HW, 256] f32.
// stats[bg*2] = mean, stats[bg*2+1] = rsqrt(var+eps)
// ---------------------------------------------------------------------------
__global__ void gn_stats_kernel(const float* __restrict__ y,
                                float* __restrict__ stats, int HW) {
    int bg = blockIdx.x;
    int b = bg >> 4, g = bg & 15;
    const float* base = y + (long long)b * HW * 256 + g * 16;
    float s = 0.f, s2 = 0.f;
    int total = HW * 16;
    for (int i = threadIdx.x; i < total; i += 256) {
        int p = i >> 4, j = i & 15;
        float v = base[(long long)p * 256 + j];
        s += v; s2 += v * v;
    }
    __shared__ float sh[512];
    sh[threadIdx.x] = s;
    sh[256 + threadIdx.x] = s2;
    __syncthreads();
    for (int off = 128; off > 0; off >>= 1) {
        if (threadIdx.x < off) {
            sh[threadIdx.x] += sh[threadIdx.x + off];
            sh[256 + threadIdx.x] += sh[256 + threadIdx.x + off];
        }
        __syncthreads();
    }
    if (threadIdx.x == 0) {
        float cnt = (float)total;
        float mu  = sh[0] / cnt;
        float var = sh[256] / cnt - mu * mu;
        stats[bg * 2]     = mu;
        stats[bg * 2 + 1] = rsqrtf(var + 1e-5f);
    }
}

// ---------------------------------------------------------------------------
// GroupNorm normalize + affine + ReLU -> padded NHWC bf16 activation.
// ---------------------------------------------------------------------------
__global__ void gn_norm_relu_kernel(const float* __restrict__ y,
                                    const float* __restrict__ stats,
                                    const float* __restrict__ gamma,
                                    const float* __restrict__ beta,
                                    unsigned short* __restrict__ actOut,
                                    int H, int W, int wshift) {
    long long i = (long long)blockIdx.x * blockDim.x + threadIdx.x;
    long long HW = (long long)H * W;
    long long total = 4LL * HW * 256;
    if (i >= total) return;
    int c = (int)(i & 255);
    long long rem = i >> 8;                 // b*HW + p
    int p = (int)(rem % HW);
    int b = (int)(rem / HW);
    int g = c >> 4;
    float mu = stats[(b * 16 + g) * 2];
    float rs = stats[(b * 16 + g) * 2 + 1];
    float v = (y[i] - mu) * rs * gamma[c] + beta[c];
    v = v > 0.f ? v : 0.f;
    int h = p >> wshift;
    int w = p & (W - 1);
    int PW = W + 2, PH = H + 2;
    size_t dst = ((size_t)(b * PH + h + 1) * PW + (w + 1)) * 256 + c;
    actOut[dst] = f2bf(v);
}

// ---------------------------------------------------------------------------
// Launcher
// ---------------------------------------------------------------------------
extern "C" void kernel_launch(void* const* d_in, const int* in_sizes, int n_in,
                              void* d_out, int out_size, void* d_ws, size_t ws_size,
                              hipStream_t stream) {
    (void)in_sizes; (void)n_in; (void)out_size; (void)ws_size;

    static const int LH[5] = {100, 50, 25, 13, 7};
    static const int LW[5] = {128, 64, 32, 16, 8};
    static const int LS[5] = {7, 6, 5, 4, 3};       // log2(W)
    const long long PADMAX = 102LL * 130;           // (H+2)*(W+2) at level 0
    const int HWMAX = 100 * 128;

    // ---- inputs (setup_inputs order) ----
    const float* feat[5];
    for (int l = 0; l < 5; ++l) feat[l] = (const float*)d_in[l];
    const float* cls_conv_w = (const float*)d_in[5];
    const float* cls_conv_b = (const float*)d_in[6];
    const float* cls_gn_g   = (const float*)d_in[7];
    const float* cls_gn_b   = (const float*)d_in[8];
    const float* cls_out_w  = (const float*)d_in[9];
    const float* cls_out_b  = (const float*)d_in[10];
    const float* reg_conv_w = (const float*)d_in[11];
    const float* reg_conv_b = (const float*)d_in[12];
    const float* reg_gn_g   = (const float*)d_in[13];
    const float* reg_gn_b   = (const float*)d_in[14];
    const float* bbox_w     = (const float*)d_in[15];
    const float* bbox_b     = (const float*)d_in[16];
    const float* ctr_w      = (const float*)d_in[17];
    const float* ctr_b      = (const float*)d_in[18];

    // ---- workspace layout ----
    size_t off = 0;
    char* base = (char*)d_ws;
    auto alloc = [&](size_t bytes) -> char* {
        char* p = base + off;
        off += (bytes + 255) & ~(size_t)255;
        return p;
    };
    unsigned short* wp_cls  = (unsigned short*)alloc((size_t)768 * 2304 * 2);
    unsigned short* wp_reg  = (unsigned short*)alloc((size_t)768 * 2304 * 2);
    unsigned short* wp_clso = (unsigned short*)alloc((size_t)80 * 2304 * 2);
    unsigned short* wp_bbox = (unsigned short*)alloc((size_t)4 * 2304 * 2);
    unsigned short* wp_ctr  = (unsigned short*)alloc((size_t)1 * 2304 * 2);
    unsigned short* actA    = (unsigned short*)alloc((size_t)4 * PADMAX * 256 * 2);
    unsigned short* actB    = (unsigned short*)alloc((size_t)4 * PADMAX * 256 * 2);
    float*          yf32    = (float*)alloc((size_t)4 * HWMAX * 256 * 4);
    float*          statsf  = (float*)alloc((size_t)128 * 4);

    // ---- pack all weights (bf16 [O][tap][ci]) ----
    {
        int t;
        t = 768 * 2304; pack_w_kernel<<<(t + 255) / 256, 256, 0, stream>>>(cls_conv_w, wp_cls, 768);
        t = 768 * 2304; pack_w_kernel<<<(t + 255) / 256, 256, 0, stream>>>(reg_conv_w, wp_reg, 768);
        t = 80  * 2304; pack_w_kernel<<<(t + 255) / 256, 256, 0, stream>>>(cls_out_w, wp_clso, 80);
        t = 4   * 2304; pack_w_kernel<<<(t + 255) / 256, 256, 0, stream>>>(bbox_w, wp_bbox, 4);
        t = 1   * 2304; pack_w_kernel<<<(t + 255) / 256, 256, 0, stream>>>(ctr_w, wp_ctr, 1);
    }

    auto launch_conv = [&](const unsigned short* act, const unsigned short* wp,
                           const float* bias, float* out, int H, int W,
                           int Cout, int relu) {
        int HW = H * W;
        int mt = (HW + 31) / 32;
        dim3 grid(4 * mt, (Cout + 127) / 128);
        conv3x3_wmma_kernel<<<grid, 128, 0, stream>>>(act, wp, bias, out, H, W, Cout, relu);
    };

    // ---- d_out offsets: 5x cls [B,H,W,80], 5x bbox [B,H,W,4], 5x ctr [B,H,W,1]
    long long cls_off[5], bbox_off[5], ctr_off[5];
    {
        long long o = 0;
        for (int l = 0; l < 5; ++l) { cls_off[l]  = o; o += 4LL * LH[l] * LW[l] * 80; }
        for (int l = 0; l < 5; ++l) { bbox_off[l] = o; o += 4LL * LH[l] * LW[l] * 4; }
        for (int l = 0; l < 5; ++l) { ctr_off[l]  = o; o += 4LL * LH[l] * LW[l] * 1; }
    }
    float* outp = (float*)d_out;

    for (int l = 0; l < 5; ++l) {
        int H = LH[l], W = LW[l], HW = H * W, ws2 = LS[l];
        long long nElem = 4LL * HW * 256;
        int eblocks = (int)((nElem + 255) / 256);

        // zero padded act buffers once per level (borders stay zero after)
        long long pad16 = 4LL * (H + 2) * (W + 2) * 256 * 2 / 16;
        int zblocks = (int)((pad16 + 255) / 256);
        zero16_kernel<<<zblocks, 256, 0, stream>>>((uint4*)actA, pad16);
        zero16_kernel<<<zblocks, 256, 0, stream>>>((uint4*)actB, pad16);

        // ================= cls tower =================
        nchw_to_nhwc_pad_kernel<<<eblocks, 256, 0, stream>>>(feat[l], actA, H, W, ws2);
        unsigned short* src = actA;
        unsigned short* dst = actB;
        for (int layer = 0; layer < 3; ++layer) {
            launch_conv(src, wp_cls + (size_t)layer * 256 * 2304,
                        cls_conv_b + layer * 256, yf32, H, W, 256, 0);
            gn_stats_kernel<<<64, 256, 0, stream>>>(yf32, statsf, HW);
            gn_norm_relu_kernel<<<eblocks, 256, 0, stream>>>(
                yf32, statsf, cls_gn_g + layer * 256, cls_gn_b + layer * 256, dst, H, W, ws2);
            unsigned short* tmp = src; src = dst; dst = tmp;
        }
        launch_conv(src, wp_clso, cls_out_b, outp + cls_off[l], H, W, 80, 0);

        // ================= reg tower =================
        nchw_to_nhwc_pad_kernel<<<eblocks, 256, 0, stream>>>(feat[l], actA, H, W, ws2);
        src = actA; dst = actB;
        for (int layer = 0; layer < 3; ++layer) {
            launch_conv(src, wp_reg + (size_t)layer * 256 * 2304,
                        reg_conv_b + layer * 256, yf32, H, W, 256, 0);
            gn_stats_kernel<<<64, 256, 0, stream>>>(yf32, statsf, HW);
            gn_norm_relu_kernel<<<eblocks, 256, 0, stream>>>(
                yf32, statsf, reg_gn_g + layer * 256, reg_gn_b + layer * 256, dst, H, W, ws2);
            unsigned short* tmp = src; src = dst; dst = tmp;
        }
        launch_conv(src, wp_bbox, bbox_b, outp + bbox_off[l], H, W, 4, 1);
        launch_conv(src, wp_ctr,  ctr_b,  outp + ctr_off[l],  H, W, 1, 0);
    }
}
